// KGramMLPSeqModel_6571299963086
// MI455X (gfx1250) — matrix operand order repaired and checked by reading, not compiled, not run on hardware
//
#include <hip/hip_runtime.h>

typedef __bf16 bf16_t;
typedef __attribute__((ext_vector_type(16))) __bf16 v16bf;
typedef __attribute__((ext_vector_type(8)))  float  v8f;

#define VOCAB 32000
#define KWIN  3
#define EMB   1024
#define HID   1024
#define SEQ   2048
#define BATCH 4
#define MROWS (SEQ * BATCH)   // 8192
#define KIN   (KWIN * EMB)    // 3072

// ---- CDNA5 async global->LDS (ASYNCcnt) path, probe-guarded ----
#if defined(__has_builtin)
#  if __has_builtin(__builtin_amdgcn_global_load_async_to_lds_b128)
#    define ASYNC_LDS 1
#  endif
#endif
#ifndef ASYNC_LDS
#  define ASYNC_LDS 0
#  warning "gfx1250 async-to-LDS builtin unavailable: using register-staged LDS copies"
#endif

__device__ __forceinline__ void wait_async0() {
#if __has_builtin(__builtin_amdgcn_s_wait_asynccnt)
  __builtin_amdgcn_s_wait_asynccnt(0);
#else
  asm volatile("s_wait_asynccnt 0x0" ::: "memory");
#endif
}

#if ASYNC_LDS
// builtin signature (from hipcc diagnostic): param0 = 'int __vector(4) __device__ *'
typedef int v4i_async __attribute__((vector_size(16)));
typedef __attribute__((address_space(1))) v4i_async* gv4i_p;  // global (__device__)
typedef __attribute__((address_space(3))) v4i_async* lv4i_p;  // LDS (__shared__)

__device__ __forceinline__ void async_cp16(const bf16_t* g, bf16_t* l) {
  __builtin_amdgcn_global_load_async_to_lds_b128(
      (gv4i_p)(const void*)g, (lv4i_p)(void*)l, 0, 0);
}
#endif

// ---------------- kernel 1: k-gram embedding gather, fp32 -> bf16 ----------------
__global__ __launch_bounds__(256) void kgram_embed_gather(
    const int* __restrict__ toks, const float* __restrict__ emb,
    bf16_t* __restrict__ X)
{
  size_t g = ((size_t)blockIdx.x * 256 + threadIdx.x) * 8;  // 8 bf16 per thread
  int m  = (int)(g / KIN);
  int k  = (int)(g % KIN);
  int s  = m >> 2, b = m & 3;          // BATCH == 4
  int kk = k >> 10, e = k & 1023;      // EMB == 1024
  int cp = s - KWIN + kk;              // context position (left-pad with token 0)
  int tok = (cp >= 0) ? toks[cp * BATCH + b] : 0;
  const float4* src = (const float4*)(emb + (size_t)tok * EMB + e);
  float4 f0 = src[0], f1 = src[1];
  union { bf16_t h[8]; uint4 u; } o;
  o.h[0] = (bf16_t)f0.x; o.h[1] = (bf16_t)f0.y; o.h[2] = (bf16_t)f0.z; o.h[3] = (bf16_t)f0.w;
  o.h[4] = (bf16_t)f1.x; o.h[5] = (bf16_t)f1.y; o.h[6] = (bf16_t)f1.z; o.h[7] = (bf16_t)f1.w;
  *(uint4*)(X + g) = o.u;
}

// ---------------- kernel 2: W (Kd x Nn, f32 row-major) -> Wt (Nn x Kd, bf16) ----------------
__global__ __launch_bounds__(256) void kgram_transpose_cvt(
    const float* __restrict__ W, bf16_t* __restrict__ Wt, int Kd, int Nn)
{
  __shared__ bf16_t tile[32][33];
  int k0 = blockIdx.x * 32, n0 = blockIdx.y * 32;
  int tx = threadIdx.x & 31, ty = threadIdx.x >> 5;   // 32 x 8
#pragma unroll
  for (int j = 0; j < 4; ++j) {
    int kr = ty + j * 8;
    tile[kr][tx] = (bf16_t)W[(size_t)(k0 + kr) * Nn + n0 + tx];
  }
  __syncthreads();
#pragma unroll
  for (int j = 0; j < 4; ++j) {
    int nr = ty + j * 8;
    Wt[(size_t)(n0 + nr) * Kd + k0 + tx] = tile[tx][nr];
  }
}

// ---------------- kernel 3/4: bf16 WMMA GEMM, 128x128 block tile, double-buffered ----------------
union Frag { v16bf v; uint4 u[2]; };
constexpr int LDK = 40;  // 32 + 8 pad (bf16 elems; 80B row stride -> conflict-free b128)

__device__ __forceinline__ void wmma_tile(
    const bf16_t* __restrict__ AsB, const bf16_t* __restrict__ BsB,
    int wm, int wn, int half, int lrow, v8f acc[4][2])
{
  // A fragment (16-bit A 16x32 layout): row = lrow, chunks at k = half*8 and 16+half*8
  Frag af[4];
#pragma unroll
  for (int mi = 0; mi < 4; ++mi) {
    const bf16_t* p = AsB + (wm * 64 + mi * 16 + lrow) * LDK + half * 8;
    af[mi].u[0] = *(const uint4*)p;
    af[mi].u[1] = *(const uint4*)(p + 16);
  }
  // B fragment (16-bit B 32x16 layout): col = lrow, chunks at k = half*8 and 16+half*8
  Frag bf_[2];
#pragma unroll
  for (int ni = 0; ni < 2; ++ni) {
    const bf16_t* p = BsB + (wn * 32 + ni * 16 + lrow) * LDK + half * 8;
    bf_[ni].u[0] = *(const uint4*)p;
    bf_[ni].u[1] = *(const uint4*)(p + 16);
  }
#pragma unroll
  for (int mi = 0; mi < 4; ++mi)
#pragma unroll
    for (int ni = 0; ni < 2; ++ni)
      acc[mi][ni] = __builtin_amdgcn_wmma_f32_16x16x32_bf16(
          false, af[mi].v, false, bf_[ni].v, (short)0, acc[mi][ni], false, false);
}

template<int EPI>   // 0: SiLU -> bf16 out   1: +bias -> f32 nontemporal out
__global__ __launch_bounds__(256) void kgram_gemm_wmma(
    const bf16_t* __restrict__ A,    // M x Kd, row-major
    const bf16_t* __restrict__ Bt,   // Nn x Kd, row-major (i.e. B transposed)
    const float*  __restrict__ bias, // Nn
    void* __restrict__ Cout,
    int M, int Nn, int Kd)
{
  __shared__ bf16_t As[2][128 * LDK];
  __shared__ bf16_t Bs[2][128 * LDK];

  const int tid  = threadIdx.x;
  const int lane = tid & 31;
  const int wid  = tid >> 5;         // 8 wave32s
  const int wm   = wid & 1;          // 2 waves along M (64 rows each)
  const int wn   = wid >> 1;         // 4 waves along N (32 cols each)
  const int half = lane >> 4;        // lane group 0/1 (K-half select)
  const int lrow = lane & 15;

  const int m0 = blockIdx.y * 128;
  const int n0 = blockIdx.x * 128;

  // cooperative tile loads: 2 threads per 32-elem K row
  const int ldr = tid >> 1;          // 0..127
  const int ldc = (tid & 1) << 4;    // 0 / 16

  const bf16_t* aG = A  + (size_t)(m0 + ldr) * Kd + ldc;
  const bf16_t* bG = Bt + (size_t)(n0 + ldr) * Kd + ldc;

  v8f acc[4][2] = {};
  const int Ktiles = Kd >> 5;

#if ASYNC_LDS
  // ---------- async DMA double-buffered pipeline ----------
  bf16_t* aL[2] = { &As[0][ldr * LDK + ldc], &As[1][ldr * LDK + ldc] };
  bf16_t* bL[2] = { &Bs[0][ldr * LDK + ldc], &Bs[1][ldr * LDK + ldc] };

  // tile 0 -> buffer 0
  async_cp16(aG,     aL[0]);
  async_cp16(aG + 8, aL[0] + 8);
  async_cp16(bG,     bL[0]);
  async_cp16(bG + 8, bL[0] + 8);

  for (int kt = 0; kt < Ktiles; ++kt) {
    const int cur = kt & 1;
    wait_async0();                   // this wave's copies for tile kt landed
    __syncthreads();                 // everyone's landed; buf[1-cur] readers done
    if (kt + 1 < Ktiles) {           // overlap next DMA with this tile's WMMAs
      const bf16_t* ap = aG + (size_t)(kt + 1) * 32;
      const bf16_t* bp = bG + (size_t)(kt + 1) * 32;
      async_cp16(ap,     aL[1 - cur]);
      async_cp16(ap + 8, aL[1 - cur] + 8);
      async_cp16(bp,     bL[1 - cur]);
      async_cp16(bp + 8, bL[1 - cur] + 8);
      if (kt + 2 < Ktiles) {
        __builtin_prefetch(aG + (size_t)(kt + 2) * 32, 0, 0);
        __builtin_prefetch(bG + (size_t)(kt + 2) * 32, 0, 0);
      }
    }
    wmma_tile(As[cur], Bs[cur], wm, wn, half, lrow, acc);
  }
#else
  // ---------- register-staged double-buffered pipeline ----------
  uint4 a0 = ((const uint4*)aG)[0], a1 = ((const uint4*)aG)[1];
  uint4 b0 = ((const uint4*)bG)[0], b1 = ((const uint4*)bG)[1];

  for (int kt = 0; kt < Ktiles; ++kt) {
    const int cur = kt & 1;
    *(uint4*)&As[cur][ldr * LDK + ldc]     = a0;
    *(uint4*)&As[cur][ldr * LDK + ldc + 8] = a1;
    *(uint4*)&Bs[cur][ldr * LDK + ldc]     = b0;
    *(uint4*)&Bs[cur][ldr * LDK + ldc + 8] = b1;
    if (kt + 1 < Ktiles) {
      const bf16_t* ap = aG + (size_t)(kt + 1) * 32;
      const bf16_t* bp = bG + (size_t)(kt + 1) * 32;
      a0 = ((const uint4*)ap)[0]; a1 = ((const uint4*)ap)[1];
      b0 = ((const uint4*)bp)[0]; b1 = ((const uint4*)bp)[1];
      if (kt + 2 < Ktiles) {
        __builtin_prefetch(aG + (size_t)(kt + 2) * 32, 0, 0);
        __builtin_prefetch(bG + (size_t)(kt + 2) * 32, 0, 0);
      }
    }
    __syncthreads();                 // single barrier per K-step (double buffer)
    wmma_tile(As[cur], Bs[cur], wm, wn, half, lrow, acc);
  }
#endif

  // epilogue: C/D layout -> VGPR i holds (row = base + half*8 + i, col = lrow)
#pragma unroll
  for (int ni = 0; ni < 2; ++ni) {
    const int col = n0 + wn * 32 + ni * 16 + lrow;
    const float bv = bias[col];
#pragma unroll
    for (int mi = 0; mi < 4; ++mi) {
      const int rbase = m0 + wm * 64 + mi * 16 + half * 8;
#pragma unroll
      for (int i = 0; i < 8; ++i) {
        float v = acc[mi][ni][i] + bv;
        size_t idx = (size_t)(rbase + i) * Nn + col;
        if (EPI == 0) {
          float sv = v / (1.0f + __expf(-v));         // SiLU
          ((bf16_t*)Cout)[idx] = (bf16_t)sv;
        } else {
          __builtin_nontemporal_store(v, (float*)Cout + idx);  // keep W2 in L2
        }
      }
    }
  }
}

// ---------------- launch ----------------
extern "C" void kernel_launch(void* const* d_in, const int* in_sizes, int n_in,
                              void* d_out, int out_size, void* d_ws, size_t ws_size,
                              hipStream_t stream) {
  const int*   toks = (const int*)d_in[0];
  const float* emb  = (const float*)d_in[1];
  const float* W1   = (const float*)d_in[2];
  const float* b1   = (const float*)d_in[3];
  const float* W2   = (const float*)d_in[4];
  const float* b2   = (const float*)d_in[5];

  char* ws = (char*)d_ws;
  bf16_t* Xbf = (bf16_t*)(ws + 0);          // 8192 x 3072  bf16 (50.3 MB)
  bf16_t* W1t = (bf16_t*)(ws + 50331648);   // 1024 x 3072  bf16 ( 6.3 MB)
  bf16_t* W2t = (bf16_t*)(ws + 56623104);   // 32000 x 1024 bf16 (65.5 MB)
  bf16_t* H   = (bf16_t*)(ws + 122159104);  // 8192 x 1024  bf16 (16.8 MB)

  // 1) gather + pack X
  kgram_embed_gather<<<dim3((MROWS * (size_t)KIN) / (8 * 256)), dim3(256), 0, stream>>>(
      toks, emb, Xbf);
  // 2) weights -> N-major bf16
  kgram_transpose_cvt<<<dim3(KIN / 32, HID / 32),  dim3(256), 0, stream>>>(W1, W1t, KIN, HID);
  kgram_transpose_cvt<<<dim3(HID / 32, VOCAB / 32), dim3(256), 0, stream>>>(W2, W2t, HID, VOCAB);
  // 3) H = silu(X @ W1 + b1)
  kgram_gemm_wmma<0><<<dim3(HID / 128, MROWS / 128), dim3(256), 0, stream>>>(
      Xbf, W1t, b1, (void*)H, MROWS, HID, KIN);
  // 4) logits = H @ W2 + b2
  kgram_gemm_wmma<1><<<dim3(VOCAB / 128, MROWS / 128), dim3(256), 0, stream>>>(
      H, W2t, b2, d_out, MROWS, VOCAB, HID);
}
